// PagedAttention_77369540870745
// MI455X (gfx1250) — matrix operand order repaired and checked by reading, not compiled, not run on hardware
//
#include <hip/hip_runtime.h>
#include <stdint.h>

// ---------------- problem constants (from reference) ----------------
#define HQ_      32
#define HKV_     8
#define GQA_     4          // HQ/HKV
#define HD_      128
#define BLKSZ_   16
#define MAXBLK_  256
#define CHUNK_   32         // tokens per wave-iteration (2 cache blocks)
#define NWAVES_  4
#define KSTRIDE_ 132        // padded LDS row stride (floats) -> bank-conflict free
#define SCALE_   0.08838834764831845f  // 128^-0.5

typedef __attribute__((ext_vector_type(16))) _Float16 v16h;
typedef __attribute__((ext_vector_type(8)))  float    v8f;
typedef __attribute__((ext_vector_type(4)))  float    v4f;

// ---------------- LDS layout ----------------
struct WaveSmem {
  float    kbuf[CHUNK_ * KSTRIDE_];   // 32 tokens x 128 f32 (padded)
  float    vbuf[CHUNK_ * KSTRIDE_];
  float    sbuf[GQA_ * CHUNK_];       // raw scores (4 rows x 32 tokens)
  _Float16 pbuf[GQA_ * CHUNK_];       // softmax'ed probs as f16
  float    alpha[GQA_];               // per-row rescale for online softmax
  float    pad_[4];
};
struct Smem {
  WaveSmem w[NWAVES_];
  float mAcc[NWAVES_][GQA_][HD_];     // per-wave unnormalized output
  float mM[NWAVES_][GQA_];
  float mL[NWAVES_][GQA_];
};

// ---------------- CDNA5 async global->LDS (ASYNCcnt) ----------------
__device__ __forceinline__ uint32_t lds_off(const void* p) {
  // flat LDS aperture: addr[31:0] is the wave-relative LDS byte offset
  return (uint32_t)(uintptr_t)p;
}
__device__ __forceinline__ void async_b128(uint32_t loff, const float* g) {
  // streaming data: 536MB > 192MB L2, zero cross-WG reuse -> non-temporal
  asm volatile("global_load_async_to_lds_b128 %0, %1, off th:TH_LOAD_NT"
               :: "v"(loff), "v"(g) : "memory");
}
__device__ __forceinline__ void wait_async_le32() {
  asm volatile("s_wait_asynccnt 0x20" ::: "memory");  // newest 32 may be pending
}
__device__ __forceinline__ void wait_async_le0() {
  asm volatile("s_wait_asynccnt 0x0" ::: "memory");
}
__device__ __forceinline__ void lds_fence() {
  asm volatile("s_wait_dscnt 0x0" ::: "memory");
}

// Stage one 32-token x 128-float slab (K or V) into LDS with exactly 32 async
// b128 ops (fixed count keeps the in-order ASYNCcnt wait math valid).
// 8 lanes cover 128B contiguous per token row. Out-of-range tokens are clamped
// to seq-1 (duplicate finite rows; masked via p=0 later).
__device__ __forceinline__ void stage_chunk(float* __restrict__ dst,
                                            const float* __restrict__ src,
                                            const int* __restrict__ bt_b,
                                            int base_tok, int seq, int kvh,
                                            int lane) {
  const int tsub = lane >> 3;        // 0..3
  const int fo   = lane & 7;         // float4 slot 0..7
  #pragma unroll
  for (int tt = 0; tt < 8; ++tt) {
    const int t = tt * 4 + tsub;     // 0..31
    int tok = base_tok + t;
    tok = (tok < seq) ? tok : (seq - 1);
    const int    blk = bt_b[tok >> 4];
    const size_t row = (((size_t)blk * BLKSZ_ + (tok & 15)) * HKV_ + kvh) * HD_;
    #pragma unroll
    for (int jj = 0; jj < 4; ++jj) {
      const int f4 = fo + jj * 8;    // float4 index 0..31 within the row
      async_b128(lds_off(dst + t * KSTRIDE_ + f4 * 4), src + row + f4 * 4);
    }
  }
}

// ---------------- kernel ----------------
__global__ void __launch_bounds__(NWAVES_ * 32)
paged_attn_decode_kernel(const float* __restrict__ q,
                         const float* __restrict__ knew,
                         const float* __restrict__ vnew,
                         const float* __restrict__ kc,
                         const float* __restrict__ vc,
                         const int*   __restrict__ btab,
                         const int*   __restrict__ seq_lens,
                         float*       __restrict__ out)
{
  extern __shared__ __align__(16) char smraw[];
  Smem* sm = (Smem*)smraw;

  const int b    = blockIdx.x;
  const int kvh  = blockIdx.y;
  const int tid  = threadIdx.x;
  const int lane = tid & 31;
  const int wave = tid >> 5;
  const bool hi  = (lane >= 16);
  const int lmod = lane & 15;
  const int seq  = seq_lens[b];
  const int* bt_b = btab + (size_t)b * MAXBLK_;

  WaveSmem* ws = &sm->w[wave];

  // ---- Q as WMMA A-fragments (M=rows: g in 0..3, rows>=4 zeroed), pre-scaled
  // A 16x32 f16, contiguous k order used here (B matches): lane m=lane%16, half j -> K = j + (hi?16:0)
  v16h aq[4];
  {
    const float* qb = q + ((size_t)b * HQ_ + (size_t)kvh * GQA_) * HD_;
    #pragma unroll
    for (int ds = 0; ds < 4; ++ds) {
      #pragma unroll
      for (int j = 0; j < 16; ++j) {
        const int K = j + (hi ? 16 : 0);
        float val = (lmod < GQA_) ? qb[lmod * HD_ + ds * 32 + K] * SCALE_ : 0.0f;
        aq[ds][j] = (_Float16)val;
      }
    }
  }

  // ---- accumulators: 8 d-tiles of 16 cols; rows 0..3 (lanes 0..15) are real
  v8f acc[8];
  #pragma unroll
  for (int i = 0; i < 8; ++i) {
    #pragma unroll
    for (int j = 0; j < 8; ++j) acc[i][j] = 0.0f;
  }
  float mrow = -3.0e38f;   // valid in lanes 0..3
  float lrow = 0.0f;

  const int nchunks = (seq + CHUNK_ - 1) / CHUNK_;
  const int last    = seq - 1;
  const float* kin  = knew + ((size_t)b * HKV_ + kvh) * HD_;
  const float* vin  = vnew + ((size_t)b * HKV_ + kvh) * HD_;

  // ---- software pipeline: K of the first chunk is prefetched here; inside the
  // loop, K of chunk c+NWAVES is prefetched into kbuf during the P*V phase
  // (kbuf is dead after softmax). In-order async completion + fixed 32-op
  // batches make "asynccnt <= 32" mean "everything but the newest batch done".
  if (wave < nchunks)
    stage_chunk(ws->kbuf, kc, bt_b, wave * CHUNK_, seq, kvh, lane);

  for (int c = wave; c < nchunks; c += NWAVES_) {
    const int base_tok = c * CHUNK_;
    int nvalid = seq - base_tok;
    if (nvalid > CHUNK_) nvalid = CHUNK_;

    stage_chunk(ws->vbuf, vc, bt_b, base_tok, seq, kvh, lane); // V(c): newest 32
    wait_async_le32();                                         // K(c) resident

    // the token written this step comes from the k input (cache not mutated)
    if (last >= base_tok && last < base_tok + CHUNK_) {
      const int t = last - base_tok;
      for (int i = lane; i < HD_; i += 32)
        ws->kbuf[t * KSTRIDE_ + i] = kin[i];
    }
    lds_fence();

    // ---- S = Q K^T : two 16-token tiles, K-dim = 128 in 4 steps of 32
    #pragma unroll
    for (int tile = 0; tile < 2; ++tile) {
      v8f cs;
      #pragma unroll
      for (int j = 0; j < 8; ++j) cs[j] = 0.0f;
      #pragma unroll
      for (int ds = 0; ds < 4; ++ds) {
        // B 32x16 f16: col n = lane%16 (= token in tile), K half-range by lane-hi
        v16h bk;
        const float* kp = &ws->kbuf[(tile * 16 + lmod) * KSTRIDE_ + ds * 32 + (hi ? 16 : 0)];
        #pragma unroll
        for (int j4 = 0; j4 < 4; ++j4) {
          v4f k4 = *(const v4f*)(kp + j4 * 4);   // 16B-aligned (132 % 4 == 0)
          bk[j4 * 4 + 0] = (_Float16)k4[0];
          bk[j4 * 4 + 1] = (_Float16)k4[1];
          bk[j4 * 4 + 2] = (_Float16)k4[2];
          bk[j4 * 4 + 3] = (_Float16)k4[3];
        }
        cs = __builtin_amdgcn_wmma_f32_16x16x32_f16(false, aq[ds], false, bk,
                                                    (short)0, cs, false, false);
      }
      // C layout: row = vgpr + 8*hi, col = lane%16 -> rows 0..3 live in !hi lanes
      if (!hi) {
        #pragma unroll
        for (int r = 0; r < GQA_; ++r)
          ws->sbuf[r * CHUNK_ + tile * 16 + lmod] = cs[r];
      }
    }
    lds_fence();

    // ---- online softmax (lane r handles row r)
    if (lane < GQA_) {
      const int r = lane;
      float rowmax = -3.0e38f;
      for (int i = 0; i < nvalid; ++i)
        rowmax = fmaxf(rowmax, ws->sbuf[r * CHUNK_ + i]);
      const float mnew = fmaxf(mrow, rowmax);
      const float al   = __expf(mrow - mnew);
      float ps = 0.0f;
      for (int i = 0; i < CHUNK_; ++i) {
        float p = (i < nvalid) ? __expf(ws->sbuf[r * CHUNK_ + i] - mnew) : 0.0f;
        ws->pbuf[r * CHUNK_ + i] = (_Float16)p;   // masked tokens -> exactly 0
        ps += p;
      }
      lrow = lrow * al + ps;
      mrow = mnew;
      ws->alpha[r] = al;
    }
    lds_fence();

    // ---- rescale accumulators (rows 0..3; other rows stay 0)
    {
      float al0 = ws->alpha[0], al1 = ws->alpha[1];
      float al2 = ws->alpha[2], al3 = ws->alpha[3];
      #pragma unroll
      for (int dt = 0; dt < 8; ++dt) {
        acc[dt][0] *= al0; acc[dt][1] *= al1;
        acc[dt][2] *= al2; acc[dt][3] *= al3;
      }
    }

    // ---- P as A-fragment (rows>=4 zero)
    v16h pa;
    #pragma unroll
    for (int j = 0; j < 16; ++j) {
      const int K = j + (hi ? 16 : 0);   // token 0..31
      pa[j] = (lmod < GQA_) ? ws->pbuf[lmod * CHUNK_ + K] : (_Float16)0.0f;
    }

    // ---- prefetch next chunk's K into kbuf (dead now), then drain V(c)
    const int cn = c + NWAVES_;
    if (cn < nchunks) {
      stage_chunk(ws->kbuf, kc, bt_b, cn * CHUNK_, seq, kvh, lane); // newest 32
      wait_async_le32();                                            // V(c) done
    } else {
      wait_async_le0();                                             // V(c) done
    }

    // the token written this step comes from the v input
    if (last >= base_tok && last < base_tok + CHUNK_) {
      const int t = last - base_tok;
      for (int i = lane; i < HD_; i += 32)
        ws->vbuf[t * KSTRIDE_ + i] = vin[i];
    }
    lds_fence();

    // ---- O += P V : 8 d-tiles of 16 columns, K-dim = 32 tokens
    #pragma unroll
    for (int dt = 0; dt < 8; ++dt) {
      v16h bv;
      #pragma unroll
      for (int j = 0; j < 16; ++j) {
        const int tok = j + (hi ? 16 : 0);
        bv[j] = (_Float16)ws->vbuf[tok * KSTRIDE_ + dt * 16 + lmod];
      }
      acc[dt] = __builtin_amdgcn_wmma_f32_16x16x32_f16(false, pa, false, bv,
                                                       (short)0, acc[dt], false, false);
    }
  }

  // ---- publish per-wave partials, then LSE-merge across the 4 waves
  if (!hi) {
    #pragma unroll
    for (int dt = 0; dt < 8; ++dt)
      #pragma unroll
      for (int r = 0; r < GQA_; ++r)
        sm->mAcc[wave][r][dt * 16 + lmod] = acc[dt][r];
  }
  if (lane < GQA_) {
    sm->mM[wave][lane] = mrow;
    sm->mL[wave][lane] = lrow;
  }
  __syncthreads();

  for (int e = tid; e < GQA_ * HD_; e += blockDim.x) {
    const int g = e >> 7;          // 0..3
    const int d = e & (HD_ - 1);
    float M = -3.0e38f;
    #pragma unroll
    for (int w = 0; w < NWAVES_; ++w) M = fmaxf(M, sm->mM[w][g]);
    float den = 0.0f, num = 0.0f;
    #pragma unroll
    for (int w = 0; w < NWAVES_; ++w) {
      const float f = __expf(sm->mM[w][g] - M);
      den += f * sm->mL[w][g];
      num += f * sm->mAcc[w][g][d];
    }
    out[(((size_t)b * HQ_) + (size_t)kvh * GQA_ + g) * HD_ + d] = num / den;
  }
}

// ---------------- launch ----------------
extern "C" void kernel_launch(void* const* d_in, const int* in_sizes, int n_in,
                              void* d_out, int out_size, void* d_ws, size_t ws_size,
                              hipStream_t stream) {
  const float* q    = (const float*)d_in[0];
  const float* knew = (const float*)d_in[1];
  const float* vnew = (const float*)d_in[2];
  const float* kc   = (const float*)d_in[3];
  const float* vc   = (const float*)d_in[4];
  // d_in[5] slot_mapping: unused — the new token is sourced directly from k/v
  const int* btab = (const int*)d_in[6];
  const int* slen = (const int*)d_in[7];
  // d_in[8] query_lens (all ones), d_in[9] is_prefill: decode path only

  const int B = in_sizes[0] / (HQ_ * HD_);
  dim3 grid(B, HKV_);
  const size_t smem = sizeof(Smem);
  paged_attn_decode_kernel<<<grid, NWAVES_ * 32, smem, stream>>>(
      q, knew, vnew, kc, vc, btab, slen, (float*)d_out);
}
